// LTCModel_29334626632063
// MI455X (gfx1250) — matrix operand order repaired and checked by reading, not compiled, not run on hardware
//
#include <hip/hip_runtime.h>
#include <hip/hip_bf16.h>

// LTC RNN on gfx1250: per-batch-row-independent recurrence -> 32 independent
// workgroups of 16 batch rows, 16 waves each owning a 16-col output tile.
// Hot loop: 10x v_wmma_f32_16x16x32_bf16 per wave per step, weights register-
// resident, h state in f32 C-fragment registers, bf16 staging through LDS.
// Double-buffered LDS staging -> single barrier per timestep.
// Sigmoid uses v_exp_f32 + v_rcp_f32 (both TRANS) to keep the serial
// elementwise tail off the expensive IEEE-div sequence.

typedef __bf16 bf16_t;
typedef __attribute__((ext_vector_type(16))) __bf16 v16bf;
typedef __attribute__((ext_vector_type(8)))  __bf16 v8bf;
typedef __attribute__((ext_vector_type(8)))  float  v8f;

#define T_STEPS 1024
#define B_TOT   512
#define D_DIM   64
#define H_DIM   256
#define O_DIM   64
#define DPH     320   // D + H

#define SRX 72    // xbuf row stride in bf16 (16x64 tile, padded)
#define SRH 264   // hbuf row stride in bf16 (16x256 tile, padded)

// Load one 16x32 bf16 A-fragment from LDS in contiguous k order (same as B):
// lane L holds row M = L&15; lanes 0-15 get K = koff..koff+15,
// lanes 16-31 get K = koff+16..+31 (koff pre-shifted by caller).
__device__ __forceinline__ v16bf lds_a_frag(const bf16_t* buf, int stride,
                                            int row, int koff) {
    const v8bf* p0 = (const v8bf*)(buf + row * stride + koff);
    const v8bf* p1 = (const v8bf*)(buf + row * stride + koff + 8);
    v8bf lo = *p0;
    v8bf hi = *p1;
    v16bf a;
#pragma unroll
    for (int i = 0; i < 8; ++i) { a[i] = lo[i]; a[8 + i] = hi[i]; }
    return a;
}

// sigmoid(x) via v_exp_f32 + v_rcp_f32 (single TRANS ops, no IEEE div chain).
__device__ __forceinline__ float fast_sigmoid(float x) {
    return __builtin_amdgcn_rcpf(1.0f + __expf(-x));
}

__global__ __launch_bounds__(512)
void ltc_persistent_kernel(const float* __restrict__ x_seq,
                           const float* __restrict__ Wf,
                           const float* __restrict__ bfv,
                           const float* __restrict__ tau,
                           const float* __restrict__ Av,
                           const float* __restrict__ Wo,
                           const float* __restrict__ bo,
                           float* __restrict__ out) {
    __shared__ bf16_t xbuf0[16 * SRX];  // x_t tile, bf16 (A cols 0..63)
    __shared__ bf16_t xbuf1[16 * SRX];
    __shared__ bf16_t hbuf0[16 * SRH];  // h tile,  bf16 (A cols 64..319)
    __shared__ bf16_t hbuf1[16 * SRH];

    const int tid  = threadIdx.x;
    const int lane = tid & 31;
    const int w    = tid >> 5;          // wave id 0..15 == N-tile id
    const int half = lane >> 4;         // 0: lanes 0-15, 1: lanes 16-31
    const int l16  = lane & 15;
    const int b0   = blockIdx.x * 16;   // batch rows [b0, b0+16)

    // Per-lane column constants for this wave's N-tile (col = 16w + l16).
    const int   ncol = w * 16 + l16;
    const float bf_n = bfv[ncol];
    const float it_n = __expf(-tau[ncol]);   // 1/exp(tau)
    const float A_n  = Av[ncol];
    const float dt   = 0.1f;

    // Register-resident B fragments: Wf^T, 10 chunks of K=32.
    // B layout (32x16 bf16): lane L -> col N = 16w + (L&15),
    // K = 32k + 16*half + {0..15} consecutive, 2 per VGPR.
    v16bf bw[10];
#pragma unroll
    for (int k = 0; k < 10; ++k) {
        const float* wp = Wf + (size_t)ncol * DPH + 32 * k + half * 16;
        v16bf b;
#pragma unroll
        for (int i = 0; i < 16; ++i) b[i] = (bf16_t)wp[i];
        bw[k] = b;
    }

    // x loader: wave w loads row (b0+w); lane loads 2 consecutive floats.
    const float* xbase =
        x_seq + (size_t)(b0 + w) * T_STEPS * D_DIM + lane * 2;
    float2 xr = *(const float2*)xbase;  // x_0

    // Init LDS: zero h in buffer 0, commit x_0 to buffer 0.
    for (int i = tid; i < 16 * SRH; i += 512) hbuf0[i] = (bf16_t)0.0f;
    xbuf0[w * SRX + lane * 2]     = (bf16_t)xr.x;
    xbuf0[w * SRX + lane * 2 + 1] = (bf16_t)xr.y;
    __syncthreads();

    v8f h = {0.f, 0.f, 0.f, 0.f, 0.f, 0.f, 0.f, 0.f};  // f32 master state

    const int arow  = l16;
    const int koff0 = half * 16;

    const bf16_t* hcur = hbuf0;
    bf16_t*       hnxt = hbuf1;
    const bf16_t* xcur = xbuf0;
    bf16_t*       xnxt = xbuf1;

    for (int t = 0; t < T_STEPS; ++t) {
        // Prefetch next x tile into registers (overlaps with WMMA work).
        if (t + 1 < T_STEPS)
            xr = *(const float2*)(xbase + (size_t)(t + 1) * D_DIM);

        v8f c = {0.f, 0.f, 0.f, 0.f, 0.f, 0.f, 0.f, 0.f};
        // K = 0..63 from x tile
#pragma unroll
        for (int k = 0; k < 2; ++k) {
            v16bf a = lds_a_frag(xcur, SRX, arow, 32 * k + koff0);
            c = __builtin_amdgcn_wmma_f32_16x16x32_bf16(
                    false, a, false, bw[k], (short)0, c, false, false);
        }
        // K = 64..319 from h tile
#pragma unroll
        for (int k = 0; k < 8; ++k) {
            v16bf a = lds_a_frag(hcur, SRH, arow, 32 * k + koff0);
            c = __builtin_amdgcn_wmma_f32_16x16x32_bf16(
                    false, a, false, bw[2 + k], (short)0, c, false, false);
        }

        // Elementwise LTC update in f32 on the C-fragment layout.
#pragma unroll
        for (int j = 0; j < 8; ++j) {
            float f  = fast_sigmoid(c[j] + bf_n);
            float hv = h[j];
            h[j] = hv + (-(it_n + f) * hv + f * A_n) * dt;
        }

        // Write next-step operands into the *other* buffer: no WAR hazard
        // with waves still reading the current buffer, so only one barrier
        // per step is needed.
        // C-fragment element j lives at row (j + 8*half), col ncol.
#pragma unroll
        for (int j = 0; j < 8; ++j)
            hnxt[(j + 8 * half) * SRH + ncol] = (bf16_t)h[j];
        xnxt[w * SRX + lane * 2]     = (bf16_t)xr.x;
        xnxt[w * SRX + lane * 2 + 1] = (bf16_t)xr.y;
        __syncthreads();  // next buffer fully visible before step t+1 reads

        // Swap buffers.
        const bf16_t* th = hcur; hcur = hnxt; hnxt = (bf16_t*)th;
        const bf16_t* tx = xcur; xcur = xnxt; xnxt = (bf16_t*)tx;
    }

    // Epilogue: out[b0:b0+16, :] = h_final @ Wo^T + bo  (waves 0..3, O=64).
    // Final h lives in hcur after the last swap.
    if (w < 4) {
        v8f c = {0.f, 0.f, 0.f, 0.f, 0.f, 0.f, 0.f, 0.f};
#pragma unroll
        for (int k = 0; k < 8; ++k) {
            v16bf a = lds_a_frag(hcur, SRH, arow, 32 * k + koff0);
            const float* wp = Wo + (size_t)ncol * H_DIM + 32 * k + half * 16;
            v16bf b;
#pragma unroll
            for (int i = 0; i < 16; ++i) b[i] = (bf16_t)wp[i];
            c = __builtin_amdgcn_wmma_f32_16x16x32_bf16(
                    false, a, false, b, (short)0, c, false, false);
        }
        const float bon = bo[ncol];
#pragma unroll
        for (int j = 0; j < 8; ++j)
            out[(size_t)(b0 + j + 8 * half) * O_DIM + ncol] = c[j] + bon;
    }
}

extern "C" void kernel_launch(void* const* d_in, const int* in_sizes, int n_in,
                              void* d_out, int out_size, void* d_ws,
                              size_t ws_size, hipStream_t stream) {
    const float* x_seq = (const float*)d_in[0];
    const float* Wf    = (const float*)d_in[1];
    const float* bfv   = (const float*)d_in[2];
    const float* tau   = (const float*)d_in[3];
    const float* Av    = (const float*)d_in[4];
    const float* Wo    = (const float*)d_in[5];
    const float* bo    = (const float*)d_in[6];
    float* out         = (float*)d_out;

    // 32 independent batch tiles of 16 rows; 16 waves (512 threads) per block.
    ltc_persistent_kernel<<<B_TOT / 16, 512, 0, stream>>>(
        x_seq, Wf, bfv, tau, Av, Wo, bo, out);
}
